// BehaviorSequenceTransformer_41308995453233
// MI455X (gfx1250) — compile-verified, matrix operand order checked by reading, not run on hardware
//
#include <hip/hip_runtime.h>
#include <hip/hip_bf16.h>
#include <stdint.h>

// Problem constants (match reference)
#define BDIM   256
#define SDIM   256
#define NPOS   (BDIM * SDIM)   // 65536 positions
#define KDIM   704             // 384 (sku) + 160 (url) + 160 (query)
#define HID    512
#define NWORD  12

typedef __attribute__((ext_vector_type(16))) __bf16          v16bf;
typedef __attribute__((ext_vector_type(8)))  float           v8f;
typedef __attribute__((ext_vector_type(8)))  unsigned short  u16x8;
typedef __attribute__((ext_vector_type(16))) unsigned short  u16x16;

__device__ __forceinline__ unsigned short f2bf(float f) {
    // round-to-nearest-even f32 -> bf16
    unsigned int u = __builtin_bit_cast(unsigned int, f);
    u += 0x7FFFu + ((u >> 16) & 1u);
    return (unsigned short)(u >> 16);
}

// Load one 16-bf16 WMMA fragment (A or B role) from an element-major row.
// p already points at row_base + k0 + (lane>>4)*8; second chunk is +16 elems.
__device__ __forceinline__ v16bf load_frag(const unsigned short* __restrict__ p) {
    u16x8 lo = *(const u16x8*)(p);
    u16x8 hi = *(const u16x8*)(p + 16);
    u16x16 t;
#pragma unroll
    for (int i = 0; i < 8; ++i) { t[i] = lo[i]; t[i + 8] = hi[i]; }
    return __builtin_bit_cast(v16bf, t);
}

// ---------------------------------------------------------------------------
// Kernel 1: pack combined weight [HID x KDIM] = [sku_w | url_w | query_w] -> bf16
// ---------------------------------------------------------------------------
__global__ void bst_pack_weights(const float* __restrict__ sku_w,
                                 const float* __restrict__ url_w,
                                 const float* __restrict__ query_w,
                                 unsigned short* __restrict__ Wbf) {
    const int n = blockIdx.x;                 // 0..511 output channel
    for (int k = threadIdx.x; k < KDIM; k += blockDim.x) {
        float v;
        if (k < 384)       v = sku_w[n * 384 + k];
        else if (k < 544)  v = url_w[n * 160 + (k - 384)];
        else               v = query_w[n * 160 + (k - 544)];
        Wbf[(size_t)n * KDIM + k] = f2bf(v);
    }
}

// ---------------------------------------------------------------------------
// Kernel 2: per-position gather + word-mean + branch-exclusive pack into X (bf16)
// One wave (32 lanes) per position; 8 positions per 256-thread block.
// Lane j owns word-mean elements {j, j+32, j+64, j+96} -> matches the k-striding
// of the packed-row write exactly (k walks lane, lane+32, ...).
// ---------------------------------------------------------------------------
__global__ __launch_bounds__(256) void bst_gather_pack(
    const float* __restrict__ event_table, const float* __restrict__ word_table,
    const float* __restrict__ sku_table,   const float* __restrict__ cat_table,
    const float* __restrict__ price_table, const float* __restrict__ url_table,
    const int* __restrict__ event_type, const int* __restrict__ sku_id,
    const int* __restrict__ url_id,     const int* __restrict__ cat_id,
    const int* __restrict__ price_id,   const int* __restrict__ word_id,
    unsigned short* __restrict__ X) {

    const int lane = threadIdx.x & 31;
    const int wave = threadIdx.x >> 5;
    const int pos  = blockIdx.x * 8 + wave;
    if (pos >= NPOS) return;

    const int et    = event_type[pos];
    const bool isSku = (et >= 1) & (et <= 3);
    const bool isUrl = (et == 4);
    const bool isQ   = (et == 5);

    float wm[4] = {0.f, 0.f, 0.f, 0.f};
    if (isSku || isQ) {
        const int* wid = word_id + (size_t)pos * NWORD;
#pragma unroll
        for (int w = 0; w < NWORD; ++w) {
            const float* row = word_table + (size_t)wid[w] * 128;
#pragma unroll
            for (int j = 0; j < 4; ++j) wm[j] += row[lane + 32 * j];
        }
        const float inv = 1.0f / (float)NWORD;
#pragma unroll
        for (int j = 0; j < 4; ++j) wm[j] *= inv;
    }

    const size_t sku  = isSku ? (size_t)sku_id[pos]   : 0;
    const size_t cat  = isSku ? (size_t)cat_id[pos]   : 0;
    const int    prc  = isSku ? price_id[pos]         : 0;
    const size_t url  = isUrl ? (size_t)url_id[pos]   : 0;

    unsigned short* xr = X + (size_t)pos * KDIM;
    for (int k = lane; k < KDIM; k += 32) {
        float v = 0.f;
        if (isSku) {
            if (k < 32)        v = event_table[et * 32 + k];
            else if (k < 160)  v = sku_table[sku * 128 + (k - 32)];
            else if (k < 224)  v = cat_table[cat * 64 + (k - 160)];
            else if (k < 256)  v = price_table[prc * 32 + (k - 224)];
            else if (k < 384)  v = wm[(k - 256) >> 5];
        } else if (isUrl) {
            if (k >= 384 && k < 512)      v = url_table[url * 128 + (k - 384)];
            else if (k >= 512 && k < 544) v = event_table[et * 32 + (k - 512)];
        } else if (isQ) {
            if (k >= 544 && k < 576)      v = event_table[et * 32 + (k - 544)];
            else if (k >= 576)            v = wm[(k - 576) >> 5];
        }
        xr[k] = f2bf(v);
    }
}

// ---------------------------------------------------------------------------
// Kernel 3: bf16 WMMA GEMM  out[M=NPOS, N=HID] = relu(X[M,K] * Wbf[N,K]^T + bias(et))
// Block: 256 threads = 8 waves; block tile 128(M) x 128(N); wave tile 32x64.
// Epilogue selects bias per row from event_type and zeroes PAD rows.
// ---------------------------------------------------------------------------
__global__ __launch_bounds__(256, 1) void bst_gemm_wmma(
    const unsigned short* __restrict__ X,
    const unsigned short* __restrict__ Wbf,
    const int* __restrict__ event_type,
    const float* __restrict__ sku_b, const float* __restrict__ url_b,
    const float* __restrict__ query_b,
    float* __restrict__ out) {

    const int lane  = threadIdx.x & 31;
    const int wave  = threadIdx.x >> 5;      // 0..7
    const int waveM = wave >> 1;             // 0..3
    const int waveN = wave & 1;              // 0..1
    const int idx   = lane & 15;             // row (A) / col (B,C) index in tile
    const int lh    = lane >> 4;             // lane-half selects K sub-range

    const int mBase = blockIdx.x * 128 + waveM * 32;
    const int nBase = blockIdx.y * 128 + waveN * 64;

    const unsigned short* aPtr[2];
    const unsigned short* bPtr[4];
#pragma unroll
    for (int mi = 0; mi < 2; ++mi)
        aPtr[mi] = X + (size_t)(mBase + mi * 16 + idx) * KDIM + lh * 8;
#pragma unroll
    for (int ni = 0; ni < 4; ++ni)
        bPtr[ni] = Wbf + (size_t)(nBase + ni * 16 + idx) * KDIM + lh * 8;

    v8f acc[2][4];
#pragma unroll
    for (int mi = 0; mi < 2; ++mi)
#pragma unroll
        for (int ni = 0; ni < 4; ++ni) acc[mi][ni] = (v8f)0.0f;

#pragma unroll 2
    for (int k0 = 0; k0 < KDIM; k0 += 32) {
        v16bf a[2], b[4];
#pragma unroll
        for (int mi = 0; mi < 2; ++mi) a[mi] = load_frag(aPtr[mi] + k0);
#pragma unroll
        for (int ni = 0; ni < 4; ++ni) b[ni] = load_frag(bPtr[ni] + k0);
#pragma unroll
        for (int mi = 0; mi < 2; ++mi)
#pragma unroll
            for (int ni = 0; ni < 4; ++ni)
                acc[mi][ni] = __builtin_amdgcn_wmma_f32_16x16x32_bf16(
                    /*neg_a=*/false, a[mi], /*neg_b=*/false, b[ni],
                    /*c_mod=*/(short)0, acc[mi][ni],
                    /*reuse_a=*/false, /*reuse_b=*/false);
    }

    // Epilogue: C layout => lane gives N=idx; VGPR r gives M row (r + 8*lh)
#pragma unroll
    for (int mi = 0; mi < 2; ++mi) {
#pragma unroll
        for (int r = 0; r < 8; ++r) {
            const int m  = mBase + mi * 16 + r + lh * 8;
            const int et = event_type[m];
            const bool sk = (et >= 1) & (et <= 3);
            const bool ur = (et == 4);
            const bool qu = (et == 5);
            const bool active = sk | ur | qu;
#pragma unroll
            for (int ni = 0; ni < 4; ++ni) {
                const int n = nBase + ni * 16 + idx;
                float bias = 0.f;
                if (sk)      bias = sku_b[n];
                else if (ur) bias = url_b[n];
                else if (qu) bias = query_b[n];
                const float v = active ? fmaxf(acc[mi][ni][r] + bias, 0.f) : 0.f;
                out[(size_t)m * HID + n] = v;
            }
        }
    }
}

// ---------------------------------------------------------------------------
// Kernel 4: padding mask (second tuple output, appended after seq_emb)
// ---------------------------------------------------------------------------
__global__ void bst_mask(const int* __restrict__ event_type,
                         float* __restrict__ mask) {
    const int i = blockIdx.x * blockDim.x + threadIdx.x;
    if (i < NPOS) mask[i] = (event_type[i] == 0) ? 1.0f : 0.0f;
}

// ---------------------------------------------------------------------------
extern "C" void kernel_launch(void* const* d_in, const int* in_sizes, int n_in,
                              void* d_out, int out_size, void* d_ws, size_t ws_size,
                              hipStream_t stream) {
    const float* event_table = (const float*)d_in[0];
    const float* word_table  = (const float*)d_in[1];
    const float* sku_table   = (const float*)d_in[2];
    const float* cat_table   = (const float*)d_in[3];
    const float* price_table = (const float*)d_in[4];
    const float* url_table   = (const float*)d_in[5];
    const float* sku_w       = (const float*)d_in[6];
    const float* sku_b       = (const float*)d_in[7];
    const float* url_w       = (const float*)d_in[8];
    const float* url_b       = (const float*)d_in[9];
    const float* query_w     = (const float*)d_in[10];
    const float* query_b     = (const float*)d_in[11];
    const int*   event_type  = (const int*)d_in[12];
    const int*   sku_id      = (const int*)d_in[13];
    const int*   url_id      = (const int*)d_in[14];
    const int*   cat_id      = (const int*)d_in[15];
    const int*   price_id    = (const int*)d_in[16];
    const int*   word_id     = (const int*)d_in[17];

    // Workspace layout: [Wbf: 512*704 bf16][X: 65536*704 bf16]
    unsigned short* Wbf = (unsigned short*)d_ws;
    size_t wbytes = ((size_t)HID * KDIM * sizeof(unsigned short) + 4095) & ~(size_t)4095;
    unsigned short* X = (unsigned short*)((char*)d_ws + wbytes);

    float* seq_emb = (float*)d_out;                       // [NPOS, HID]
    float* mask    = seq_emb + (size_t)NPOS * HID;        // [NPOS]

    bst_pack_weights<<<HID, 256, 0, stream>>>(sku_w, url_w, query_w, Wbf);

    bst_gather_pack<<<NPOS / 8, 256, 0, stream>>>(
        event_table, word_table, sku_table, cat_table, price_table, url_table,
        event_type, sku_id, url_id, cat_id, price_id, word_id, X);

    dim3 grid(NPOS / 128, HID / 128);
    bst_gemm_wmma<<<grid, 256, 0, stream>>>(X, Wbf, event_type,
                                            sku_b, url_b, query_b, seq_emb);

    bst_mask<<<NPOS / 256, 256, 0, stream>>>(event_type, mask);
}
